// BinarizedFCLayer_19164144075028
// MI455X (gfx1250) — matrix operand order repaired and checked by reading, not compiled, not run on hardware
//
#include <hip/hip_runtime.h>

typedef __bf16 bf16_t;
typedef __attribute__((ext_vector_type(16))) __bf16 v16bf;
typedef __attribute__((ext_vector_type(8)))  __bf16 v8bf;
typedef __attribute__((ext_vector_type(8)))  float  v8f;
typedef __attribute__((ext_vector_type(4)))  float  f32x4;
typedef __attribute__((ext_vector_type(4)))  unsigned int u32x4;
typedef __attribute__((ext_vector_type(8)))  unsigned int u32x8;

#define K_DIM 2048
#define N_DIM 2048

__device__ __forceinline__ bf16_t bin_pm1(float f) {
    // sign(f) with sign(0)->+1, exact in bf16: 0x3F80 = +1.0, OR in sign bit.
    unsigned int u = __float_as_uint(f);
    unsigned short s = (unsigned short)(0x3F80u | ((u >> 16) & 0x8000u));
    return __builtin_bit_cast(bf16_t, s);
}

// ---------------- Pre-pass 1: X f32 -> bf16 (RNE), 8 elems/thread ----------
__global__ __launch_bounds__(256)
void cvt_x_bf16(const float* __restrict__ in, bf16_t* __restrict__ out, int n8) {
    int t = blockIdx.x * blockDim.x + threadIdx.x;
    if (t >= n8) return;
    const f32x4* p = (const f32x4*)in + (size_t)2 * t;
    f32x4 x0 = p[0];
    f32x4 x1 = p[1];
    v8bf o;
#pragma unroll
    for (int e = 0; e < 4; ++e) {
        o[e]     = (bf16_t)x0[e];
        o[4 + e] = (bf16_t)x1[e];
    }
    *((v8bf*)out + t) = o;
}

// ---------------- Pre-pass 2: W f32 -> sign(W) in bf16 (+/-1) --------------
__global__ __launch_bounds__(256)
void binz_w_bf16(const float* __restrict__ in, bf16_t* __restrict__ out, int n8) {
    int t = blockIdx.x * blockDim.x + threadIdx.x;
    if (t >= n8) return;
    const f32x4* p = (const f32x4*)in + (size_t)2 * t;
    f32x4 x0 = p[0];
    f32x4 x1 = p[1];
    v8bf o;
#pragma unroll
    for (int e = 0; e < 4; ++e) {
        o[e]     = bin_pm1(x0[e]);
        o[4 + e] = bin_pm1(x1[e]);
    }
    *((v8bf*)out + t) = o;
}

// ---------------- Main GEMM: pure bf16 loads + WMMA ------------------------
// WG = 8 waves (2 M x 4 N); wave tile 64x64 as 4x4 blocks of 16x16.
// Inner loop per K-step of 32: 16 global_load_b128 + 16 v_wmma, no cvt VALU.
__global__ __launch_bounds__(256)
void bnn_gemm_bf16(const bf16_t* __restrict__ Xb,  // [M, K] bf16
                   const bf16_t* __restrict__ Wb,  // [N, K] bf16 (+/-1)
                   float* __restrict__ C,          // [M, N] f32
                   int M) {
    const int lane = threadIdx.x & 31;
    const int wave = threadIdx.x >> 5;
    const int mrow = lane & 15;   // A row / B,C column owned by this lane
    const int sel  = lane >> 4;   // K/M half select per ISA 7.12.2

    const int wm0 = blockIdx.x * 128 + (wave >> 2) * 64;
    const int wn0 = blockIdx.y * 256 + (wave & 3) * 64;
    if (wm0 >= M) return;

    v8f acc[4][4];
#pragma unroll
    for (int i = 0; i < 4; ++i)
#pragma unroll
        for (int j = 0; j < 4; ++j)
            acc[i][j] = (v8f){0.f, 0.f, 0.f, 0.f, 0.f, 0.f, 0.f, 0.f};

    const bf16_t* arow[4];
#pragma unroll
    for (int i = 0; i < 4; ++i)
        arow[i] = Xb + (size_t)(wm0 + i * 16 + mrow) * K_DIM;
    const bf16_t* brow[4];
#pragma unroll
    for (int j = 0; j < 4; ++j)
        brow[j] = Wb + (size_t)(wn0 + j * 16 + mrow) * K_DIM;

    for (int k0 = 0; k0 < K_DIM; k0 += 32) {
        // A fragment: elems 0..7 at K=k0+sel*8, elems 8..15 at K=k0+16+sel*8.
        v16bf a[4];
#pragma unroll
        for (int i = 0; i < 4; ++i) {
            const u32x4* p = (const u32x4*)(arow[i] + k0 + sel * 8); // 16B aligned
            u32x4 lo = p[0];       // 8 bf16
            u32x4 hi = p[2];       // +16 bf16 = +32B
            u32x8 m = __builtin_shufflevector(lo, hi, 0, 1, 2, 3, 4, 5, 6, 7);
            a[i] = __builtin_bit_cast(v16bf, m);
        }
        // B fragment: elems 0..7 at K=k0+sel*8, elems 8..15 at K=k0+16+sel*8.
        v16bf b[4];
#pragma unroll
        for (int j = 0; j < 4; ++j) {
            const u32x4* q = (const u32x4*)(brow[j] + k0 + sel * 8); // 16B aligned
            u32x4 lo = q[0];
            u32x4 hi = q[2];
            u32x8 m = __builtin_shufflevector(lo, hi, 0, 1, 2, 3, 4, 5, 6, 7);
            b[j] = __builtin_bit_cast(v16bf, m);
        }
        // 16 WMMAs
#pragma unroll
        for (int i = 0; i < 4; ++i)
#pragma unroll
            for (int j = 0; j < 4; ++j)
                acc[i][j] = __builtin_amdgcn_wmma_f32_16x16x32_bf16(
                    false, a[i], false, b[j], (short)0, acc[i][j], false, false);
    }

#pragma unroll
    for (int i = 0; i < 4; ++i)
#pragma unroll
        for (int j = 0; j < 4; ++j) {
            const int col = wn0 + j * 16 + mrow;
#pragma unroll
            for (int r = 0; r < 8; ++r) {
                const int row = wm0 + i * 16 + sel * 8 + r;
                C[(size_t)row * N_DIM + col] = acc[i][j][r];
            }
        }
}

// ---------------- Fallback: fused f32->bf16 GEMM (round-1 kernel) ----------
__global__ __launch_bounds__(256)
void bnn_gemm_fused(const float* __restrict__ X, const float* __restrict__ W,
                    float* __restrict__ C, int M) {
    const int lane = threadIdx.x & 31;
    const int wave = threadIdx.x >> 5;
    const int mrow = lane & 15;
    const int sel  = lane >> 4;

    const int wm0 = blockIdx.x * 128 + (wave >> 2) * 64;
    const int wn0 = blockIdx.y * 128 + (wave & 3) * 32;
    if (wm0 >= M) return;

    v8f acc[4][2];
#pragma unroll
    for (int i = 0; i < 4; ++i)
#pragma unroll
        for (int j = 0; j < 2; ++j)
            acc[i][j] = (v8f){0.f, 0.f, 0.f, 0.f, 0.f, 0.f, 0.f, 0.f};

    const float* arow[4];
#pragma unroll
    for (int i = 0; i < 4; ++i)
        arow[i] = X + (size_t)(wm0 + i * 16 + mrow) * K_DIM;
    const float* brow[2];
#pragma unroll
    for (int j = 0; j < 2; ++j)
        brow[j] = W + (size_t)(wn0 + j * 16 + mrow) * K_DIM;

    for (int k0 = 0; k0 < K_DIM; k0 += 32) {
        v16bf a[4];
#pragma unroll
        for (int i = 0; i < 4; ++i) {
            const float* p = arow[i] + k0 + sel * 8;
            f32x4 x0 = *(const f32x4*)(p);
            f32x4 x1 = *(const f32x4*)(p + 4);
            f32x4 x2 = *(const f32x4*)(p + 16);
            f32x4 x3 = *(const f32x4*)(p + 20);
#pragma unroll
            for (int e = 0; e < 4; ++e) {
                a[i][e]      = (bf16_t)x0[e];
                a[i][4 + e]  = (bf16_t)x1[e];
                a[i][8 + e]  = (bf16_t)x2[e];
                a[i][12 + e] = (bf16_t)x3[e];
            }
        }
        v16bf b[2];
#pragma unroll
        for (int j = 0; j < 2; ++j) {
            const float* q = brow[j] + k0 + sel * 8;
            f32x4 w0 = *(const f32x4*)(q);
            f32x4 w1 = *(const f32x4*)(q + 4);
            f32x4 w2 = *(const f32x4*)(q + 16);
            f32x4 w3 = *(const f32x4*)(q + 20);
#pragma unroll
            for (int e = 0; e < 4; ++e) {
                b[j][e]      = bin_pm1(w0[e]);
                b[j][4 + e]  = bin_pm1(w1[e]);
                b[j][8 + e]  = bin_pm1(w2[e]);
                b[j][12 + e] = bin_pm1(w3[e]);
            }
        }
#pragma unroll
        for (int i = 0; i < 4; ++i)
#pragma unroll
            for (int j = 0; j < 2; ++j)
                acc[i][j] = __builtin_amdgcn_wmma_f32_16x16x32_bf16(
                    false, a[i], false, b[j], (short)0, acc[i][j], false, false);
    }

#pragma unroll
    for (int i = 0; i < 4; ++i)
#pragma unroll
        for (int j = 0; j < 2; ++j) {
            const int col = wn0 + j * 16 + mrow;
#pragma unroll
            for (int r = 0; r < 8; ++r) {
                const int row = wm0 + i * 16 + sel * 8 + r;
                C[(size_t)row * N_DIM + col] = acc[i][j][r];
            }
        }
}

extern "C" void kernel_launch(void* const* d_in, const int* in_sizes, int n_in,
                              void* d_out, int out_size, void* d_ws, size_t ws_size,
                              hipStream_t stream) {
    const float* X = (const float*)d_in[0];
    const float* W = (const float*)d_in[1];
    float* C = (float*)d_out;
    const int M = in_sizes[0] / K_DIM;  // 8192

    const size_t xbytes = (size_t)M * K_DIM * sizeof(bf16_t);      // 32 MB
    const size_t wbytes = (size_t)N_DIM * K_DIM * sizeof(bf16_t);  //  8 MB

    if (ws_size >= xbytes + wbytes) {
        bf16_t* Xb = (bf16_t*)d_ws;
        bf16_t* Wb = (bf16_t*)((char*)d_ws + xbytes);
        const int nx8 = (M * K_DIM) / 8;
        const int nw8 = (N_DIM * K_DIM) / 8;
        cvt_x_bf16<<<dim3((nx8 + 255) / 256), dim3(256), 0, stream>>>(X, Xb, nx8);
        binz_w_bf16<<<dim3((nw8 + 255) / 256), dim3(256), 0, stream>>>(W, Wb, nw8);
        bnn_gemm_bf16<<<dim3(M / 128, N_DIM / 256), dim3(256), 0, stream>>>(Xb, Wb, C, M);
    } else {
        bnn_gemm_fused<<<dim3(M / 128, N_DIM / 128), dim3(256), 0, stream>>>(X, W, C, M);
    }
}
